// QwenAttention_9345848836836
// MI455X (gfx1250) — matrix operand order repaired and checked by reading, not compiled, not run on hardware
//
#include <hip/hip_runtime.h>
#include <hip/hip_bf16.h>
#include <stdint.h>

// ---------------------------------------------------------------------------
// Types for CDNA5 WMMA (wave32): v_wmma_f32_16x16x32_bf16
// ---------------------------------------------------------------------------
typedef __attribute__((ext_vector_type(16))) __bf16 v16bf;
typedef __attribute__((ext_vector_type(8)))  float  v8f;

#define N_HEADS   28
#define N_KV      4
#define HEAD_DIM  128
#define SEQ       2048
#define HID       3584
#define BATCH     2
#define MROWS     (BATCH * SEQ)           // 4096
#define ATT_SCALE 0.08838834764831845f    // 1/sqrt(128)

// ---------------------------------------------------------------------------
// bf16 <-> f32 helpers (round-to-nearest-even)
// ---------------------------------------------------------------------------
__device__ __forceinline__ __bf16 f2bf(float f) {
  unsigned u = __builtin_bit_cast(unsigned, f);
  unsigned r = u + 0x7FFFu + ((u >> 16) & 1u);
  unsigned short h = (unsigned short)(r >> 16);
  return __builtin_bit_cast(__bf16, h);
}
__device__ __forceinline__ float bf2f(__bf16 b) {
  unsigned short h = __builtin_bit_cast(unsigned short, b);
  unsigned u = ((unsigned)h) << 16;
  return __builtin_bit_cast(float, u);
}

// ---------------------------------------------------------------------------
// CDNA5 async global->LDS copies (ASYNCcnt path, CDNA5 ISA)
// 16B per lane per instruction; completion via s_wait_asynccnt (in-order).
// ---------------------------------------------------------------------------
__device__ __forceinline__ uint32_t lds_off_of(const void* p) {
  // generic -> LDS addrspace cast yields the 32-bit LDS byte offset
  return (uint32_t)(uintptr_t)(__attribute__((address_space(3))) const char*)p;
}
__device__ __forceinline__ void async_copy_b128(void* lds_dst, const void* gsrc) {
  asm volatile("global_load_async_to_lds_b128 %0, %1, off"
               :: "v"(lds_off_of(lds_dst)), "v"(gsrc)
               : "memory");
}
template <int N>
__device__ __forceinline__ void wait_asynccnt() {
  asm volatile("s_wait_asynccnt %0" :: "n"(N) : "memory");
}

// ---------------------------------------------------------------------------
// Fragment loaders (wave32 layouts per CDNA5 ISA §7.12.2)
//  A 16x32 bf16: lane l -> row (l%16); elems e: K = (e/8)*16 + (l/16)*8 + e%8
//  B 32x16 bf16: lane l -> col (l%16); elems e: K = (e/8)*16 + (l/16)*8 + e%8
// rowPtr points at K=0 of this lane's row/column for the current k-step.
// ---------------------------------------------------------------------------
__device__ __forceinline__ v16bf load_a_frag(const __bf16* rowPtr, int half) {
  union { v16bf v; uint4 q[2]; } u;
  u.q[0] = *(const uint4*)(rowPtr + half * 8);
  u.q[1] = *(const uint4*)(rowPtr + 16 + half * 8);
  return u.v;
}
__device__ __forceinline__ v16bf load_b_frag(const __bf16* rowPtr, int half) {
  union { v16bf v; uint4 q[2]; } u;
  u.q[0] = *(const uint4*)(rowPtr + half * 8);
  u.q[1] = *(const uint4*)(rowPtr + 16 + half * 8);
  return u.v;
}
__device__ __forceinline__ v8f wmma_bf16(v16bf a, v16bf b, v8f c) {
  return __builtin_amdgcn_wmma_f32_16x16x32_bf16(false, a, false, b,
                                                 (short)0, c, false, false);
}

// ---------------------------------------------------------------------------
// Kernel: fp32 -> bf16 elementwise convert
// ---------------------------------------------------------------------------
__global__ void convert_bf16_kernel(const float* __restrict__ x,
                                    __bf16* __restrict__ y, size_t n) {
  size_t i = (size_t)blockIdx.x * blockDim.x + threadIdx.x;
  size_t stride = (size_t)gridDim.x * blockDim.x;
  for (; i < n; i += stride) y[i] = f2bf(x[i]);
}

// ---------------------------------------------------------------------------
// Kernel: W[K][N] fp32 -> Wt[N][K] bf16 (32x32 LDS tile transpose)
// ---------------------------------------------------------------------------
__global__ __launch_bounds__(256)
void transpose_bf16_kernel(const float* __restrict__ W, __bf16* __restrict__ Wt,
                           int K, int N) {
  __shared__ float tile[32][33];
  int tx = threadIdx.x, ty = threadIdx.y;   // 32 x 8
  int n = blockIdx.x * 32 + tx;
  int kBase = blockIdx.y * 32;
#pragma unroll
  for (int i = 0; i < 32; i += 8)
    tile[ty + i][tx] = W[(size_t)(kBase + ty + i) * N + n];
  __syncthreads();
  int k = kBase + tx;
#pragma unroll
  for (int i = 0; i < 32; i += 8) {
    int n2 = blockIdx.x * 32 + ty + i;
    Wt[(size_t)n2 * K + k] = f2bf(tile[tx][ty + i]);
  }
}

// ---------------------------------------------------------------------------
// Kernel: tiled bf16 WMMA GEMM.  C[M,N] = A[M,K] * Bt[N,K]^T
// 256 threads = 8 waves; block tile 128x128; each wave -> 16 rows x 128 cols.
// Double-buffered LDS; k+1 slab copied with global_load_async_to_lds_b128
// while WMMAs consume the current slab (4 async ops in flight per stage).
// mode 0: write fp32 row-major to outF[M][N]
// mode 1: write bf16 to outB in [b, head, s, d] layout (head = n/128)
// ---------------------------------------------------------------------------
__global__ __launch_bounds__(256)
void gemm_wmma_kernel(const __bf16* __restrict__ A, const __bf16* __restrict__ Bt,
                      int M, int N, int K, int mode,
                      float* __restrict__ outF, __bf16* __restrict__ outB,
                      int nHeadsOut) {
  __shared__ __align__(16) __bf16 As[2][128 * 32];
  __shared__ __align__(16) __bf16 Bs[2][128 * 32];
  const int tid = threadIdx.x;
  const int lane = tid & 31, wave = tid >> 5;
  const int half = lane >> 4, l16 = lane & 15;
  const int n0 = blockIdx.x * 128, m0 = blockIdx.y * 128;

  // Each thread copies two 16B chunks of A and two of B per stage (4 async ops)
  const int rowS = tid >> 2;               // 0..63  (chunk rows, +64 for c=1)
  const int offS = (tid & 3) * 8;          // bf16 offset within 32-wide k slab

  auto stage = [&](int buf, int kb) {
#pragma unroll
    for (int c = 0; c < 2; ++c) {
      int row = rowS + c * 64;
      async_copy_b128(As[buf] + row * 32 + offS,
                      A + (size_t)(m0 + row) * K + kb * 32 + offS);
      async_copy_b128(Bs[buf] + row * 32 + offS,
                      Bt + (size_t)(n0 + row) * K + kb * 32 + offS);
    }
  };

  v8f acc[8] = {};
  const int steps = K >> 5;
  stage(0, 0);
  for (int kb = 0; kb < steps; ++kb) {
    const int cur = kb & 1;
    if (kb + 1 < steps) {
      stage(cur ^ 1, kb + 1);
      wait_asynccnt<4>();   // previous stage's 4 copies (in-order) have landed
    } else {
      wait_asynccnt<0>();
    }
    __syncthreads();

    v16bf a = load_a_frag(As[cur] + (wave * 16 + l16) * 32, half);
    v16bf bfr[8];
#pragma unroll
    for (int j = 0; j < 8; ++j)
      bfr[j] = load_b_frag(Bs[cur] + (j * 16 + l16) * 32, half);
#pragma unroll
    for (int j = 0; j < 8; ++j)
      acc[j] = wmma_bf16(a, bfr[j], acc[j]);
    __syncthreads();
  }

  // Epilogue.  C layout: lane l, vgpr r -> row = r + 8*(l/16), col = l%16.
#pragma unroll
  for (int j = 0; j < 8; ++j) {
#pragma unroll
    for (int r = 0; r < 8; ++r) {
      int mg = m0 + wave * 16 + r + 8 * half;
      int ng = n0 + j * 16 + l16;
      float v = acc[j][r];
      if (mode == 0) {
        outF[(size_t)mg * N + ng] = v;
      } else {
        int b = mg / SEQ, s = mg - b * SEQ;
        int h = ng >> 7, d = ng & 127;
        outB[(((size_t)(b * nHeadsOut + h)) * SEQ + s) * HEAD_DIM + d] = f2bf(v);
      }
    }
  }
}

// ---------------------------------------------------------------------------
// RoPE (GPT-J interleaved sincos, split-half variant matching the reference)
// X is bf16 [rows][128]; position p = row % SEQ; thread handles dims j, j+64.
// ---------------------------------------------------------------------------
__device__ __forceinline__ float sincos_col(int p, int c) {
  int i = c >> 1;
  float invf = __powf(1.0e6f, -(float)i / 64.0f);
  float a = (float)p * invf;
  return (c & 1) ? __cosf(a) : __sinf(a);
}
__global__ void rope_kernel(__bf16* __restrict__ X, size_t nRows) {
  size_t idx = (size_t)blockIdx.x * blockDim.x + threadIdx.x;
  if (idx >= nRows * 64) return;
  int j = (int)(idx & 63);
  size_t row = idx >> 6;
  int p = (int)(row % SEQ);
  __bf16* x = X + row * HEAD_DIM;
  float xj = bf2f(x[j]), xk = bf2f(x[j + 64]);
  float cv = sincos_col(p, 64 + j);  // cos_v(j) == cos_v(j+64)
  float sv = sincos_col(p, j);       // sin_v(j) == sin_v(j+64)
  // rotate_half: rh[j] = -x[j+64]; rh[j+64] = x[j]
  x[j]      = f2bf(xj * cv - xk * sv);
  x[j + 64] = f2bf(xk * cv + xj * sv);
}

// ---------------------------------------------------------------------------
// Flash attention (causal, GQA).  One block = (b, head, 128 q rows).
// 8 waves; each wave owns 16 q rows (Q kept in 4 A-fragments).
// kv processed in blocks of 32: K staged [32][128] via async-to-LDS copies,
// V staged transposed [128][32] so PV B-fragments are K-contiguous.
// Softmax fully in registers (row lives in one 16-lane half of the C layout
// -> __shfl_xor reductions over lane masks 1..8).
// ---------------------------------------------------------------------------
__global__ __launch_bounds__(256)
void flash_attn_kernel(const __bf16* __restrict__ Qh,  // [B,28,S,128]
                       const __bf16* __restrict__ Kh,  // [B,4,S,128]
                       const __bf16* __restrict__ Vh,  // [B,4,S,128]
                       __bf16* __restrict__ O) {       // [B*S, 3584]
  __shared__ __align__(16) __bf16 Ks[32 * 128];
  __shared__ __align__(16) __bf16 Vts[128 * 32];
  __shared__ __align__(16) __bf16 Ps[8][16 * 32];

  const int tid = threadIdx.x;
  const int lane = tid & 31, wave = tid >> 5;
  const int half = lane >> 4, l16 = lane & 15;

  int blk = blockIdx.x;
  int qb = blk & 15;                 // S/128 = 16 q-blocks
  int h = (blk >> 4) % N_HEADS;
  int b = blk / (16 * N_HEADS);
  int hkv = h & (N_KV - 1);          // jnp.tile block-repeat -> h % 4

  const __bf16* Qbase =
      Qh + (((size_t)(b * N_HEADS + h)) * SEQ + qb * 128 + wave * 16) * HEAD_DIM;
  const __bf16* Kbase = Kh + ((size_t)(b * N_KV + hkv)) * SEQ * HEAD_DIM;
  const __bf16* Vbase = Vh + ((size_t)(b * N_KV + hkv)) * SEQ * HEAD_DIM;

  // Q fragments: 16 rows x 128, 4 k-steps of 32
  v16bf qf[4];
#pragma unroll
  for (int k = 0; k < 4; ++k)
    qf[k] = load_a_frag(Qbase + l16 * HEAD_DIM + k * 32, half);

  v8f o[8] = {};
  float mrun[8], lrun[8];
#pragma unroll
  for (int r = 0; r < 8; ++r) { mrun[r] = -1.0e30f; lrun[r] = 0.0f; }

  const int q0 = qb * 128;
  const int nkv = (q0 + 128) >> 5;   // causal: only kv blocks <= q range

  for (int kv = 0; kv < nkv; ++kv) {
    const int s0 = kv * 32;
    // Stage K rows (32 x 256B) via async copies; V transposed via VGPRs.
#pragma unroll
    for (int c = 0; c < 2; ++c) {
      int chunk = tid + c * 256;
      int row = chunk >> 4, off = (chunk & 15) * 8;
      async_copy_b128(Ks + row * 128 + off,
                      Kbase + (size_t)(s0 + row) * HEAD_DIM + off);
      const __bf16* vsrc = Vbase + (size_t)(s0 + row) * HEAD_DIM + off;
#pragma unroll
      for (int e = 0; e < 8; ++e) Vts[(off + e) * 32 + row] = vsrc[e];
    }
    wait_asynccnt<0>();
    __syncthreads();

    // Scores: S(16x32) = Q(16x128) . K^T  (B-frag rows are K rows of LDS)
    v8f sfr[2] = {};
#pragma unroll
    for (int nt = 0; nt < 2; ++nt) {
      v16bf kb4[4];
#pragma unroll
      for (int k = 0; k < 4; ++k)
        kb4[k] = load_b_frag(Ks + (nt * 16 + l16) * 128 + k * 32, half);
#pragma unroll
      for (int k = 0; k < 4; ++k)
        sfr[nt] = wmma_bf16(qf[k], kb4[k], sfr[nt]);
    }

    // Online softmax per row (row = r + 8*half; cols striped over 16 lanes)
    float pr0[8], pr1[8];
#pragma unroll
    for (int r = 0; r < 8; ++r) {
      int qrow = q0 + wave * 16 + r + 8 * half;
      float v0 = sfr[0][r] * ATT_SCALE;
      float v1 = sfr[1][r] * ATT_SCALE;
      if (s0 + l16 > qrow)      v0 = -1.0e30f;
      if (s0 + 16 + l16 > qrow) v1 = -1.0e30f;
      float mx = fmaxf(v0, v1);
#pragma unroll
      for (int sh = 1; sh < 16; sh <<= 1) mx = fmaxf(mx, __shfl_xor(mx, sh, 32));
      float mnew = fmaxf(mrun[r], mx);
      float rescale = __expf(mrun[r] - mnew);
      float p0 = __expf(v0 - mnew), p1 = __expf(v1 - mnew);
      float sum = p0 + p1;
#pragma unroll
      for (int sh = 1; sh < 16; sh <<= 1) sum += __shfl_xor(sum, sh, 32);
      lrun[r] = lrun[r] * rescale + sum;
      mrun[r] = mnew;
      pr0[r] = p0; pr1[r] = p1;
#pragma unroll
      for (int j = 0; j < 8; ++j) o[j][r] *= rescale;
    }

    // Relayout P (C layout) -> A layout via per-wave LDS scratch
    __bf16* P = &Ps[wave][0];
#pragma unroll
    for (int r = 0; r < 8; ++r) {
      int m = r + 8 * half;
      P[m * 32 + l16]      = f2bf(pr0[r]);
      P[m * 32 + 16 + l16] = f2bf(pr1[r]);
    }
    v16bf pa = load_a_frag(P + l16 * 32, half);
    v16bf vfr[8];
#pragma unroll
    for (int j = 0; j < 8; ++j)
      vfr[j] = load_b_frag(Vts + (j * 16 + l16) * 32, half);
#pragma unroll
    for (int j = 0; j < 8; ++j)
      o[j] = wmma_bf16(pa, vfr[j], o[j]);
    __syncthreads();
  }

  // Normalize and write merged-head bf16 rows [B*S][3584]
#pragma unroll
  for (int r = 0; r < 8; ++r) {
    float inv = 1.0f / lrun[r];
    int qrow = q0 + wave * 16 + r + 8 * half;
    size_t rowoff = ((size_t)b * SEQ + qrow) * (size_t)HID + (size_t)h * HEAD_DIM;
#pragma unroll
    for (int j = 0; j < 8; ++j)
      O[rowoff + j * 16 + l16] = f2bf(o[j][r] * inv);
  }
}

// ---------------------------------------------------------------------------
// Host-side orchestration
// ---------------------------------------------------------------------------
static inline char* ws_carve(char*& p, size_t bytes) {
  char* r = p;
  p += (bytes + 255) & ~(size_t)255;
  return r;
}

extern "C" void kernel_launch(void* const* d_in, const int* in_sizes, int n_in,
                              void* d_out, int out_size, void* d_ws, size_t ws_size,
                              hipStream_t stream) {
  const float* hs = (const float*)d_in[0];  // [B,S,H]
  const float* wq = (const float*)d_in[1];  // [H,3584]
  const float* wk = (const float*)d_in[2];  // [H,512]
  const float* wv = (const float*)d_in[3];  // [H,512]
  const float* wo = (const float*)d_in[4];  // [3584,H]
  (void)in_sizes; (void)n_in; (void)out_size; (void)ws_size;

  char* p = (char*)d_ws;
  __bf16* Xbf = (__bf16*)ws_carve(p, (size_t)MROWS * HID * 2);
  __bf16* WqT = (__bf16*)ws_carve(p, (size_t)HID * 3584 * 2);
  __bf16* WkT = (__bf16*)ws_carve(p, (size_t)HID * 512 * 2);
  __bf16* WvT = (__bf16*)ws_carve(p, (size_t)HID * 512 * 2);
  __bf16* WoT = (__bf16*)ws_carve(p, (size_t)HID * 3584 * 2);
  __bf16* Qh  = (__bf16*)ws_carve(p, (size_t)BATCH * N_HEADS * SEQ * HEAD_DIM * 2);
  __bf16* Kh  = (__bf16*)ws_carve(p, (size_t)BATCH * N_KV * SEQ * HEAD_DIM * 2);
  __bf16* Vh  = (__bf16*)ws_carve(p, (size_t)BATCH * N_KV * SEQ * HEAD_DIM * 2);
  __bf16* AO  = (__bf16*)ws_carve(p, (size_t)MROWS * HID * 2);

  // 1) Activations -> bf16; weights -> transposed bf16
  convert_bf16_kernel<<<4096, 256, 0, stream>>>(hs, Xbf, (size_t)MROWS * HID);
  {
    dim3 blk(32, 8);
    transpose_bf16_kernel<<<dim3(3584 / 32, HID / 32), blk, 0, stream>>>(wq, WqT, HID, 3584);
    transpose_bf16_kernel<<<dim3(512 / 32,  HID / 32), blk, 0, stream>>>(wk, WkT, HID, 512);
    transpose_bf16_kernel<<<dim3(512 / 32,  HID / 32), blk, 0, stream>>>(wv, WvT, HID, 512);
    transpose_bf16_kernel<<<dim3(3584 / 32, HID / 32), blk, 0, stream>>>(wo, WoT, HID, 3584);
  }

  // 2) QKV projections (WMMA GEMM, head-layout epilogue)
  gemm_wmma_kernel<<<dim3(3584 / 128, MROWS / 128), 256, 0, stream>>>(
      Xbf, WqT, MROWS, 3584, HID, 1, nullptr, Qh, N_HEADS);
  gemm_wmma_kernel<<<dim3(512 / 128, MROWS / 128), 256, 0, stream>>>(
      Xbf, WkT, MROWS, 512, HID, 1, nullptr, Kh, N_KV);
  gemm_wmma_kernel<<<dim3(512 / 128, MROWS / 128), 256, 0, stream>>>(
      Xbf, WvT, MROWS, 512, HID, 1, nullptr, Vh, N_KV);

  // 3) RoPE on Q and K (in place)
  {
    size_t qrows = (size_t)BATCH * N_HEADS * SEQ;
    size_t krows = (size_t)BATCH * N_KV * SEQ;
    rope_kernel<<<(unsigned)((qrows * 64 + 255) / 256), 256, 0, stream>>>(Qh, qrows);
    rope_kernel<<<(unsigned)((krows * 64 + 255) / 256), 256, 0, stream>>>(Kh, krows);
  }

  // 4) Causal GQA flash attention -> merged heads bf16 [B*S][3584]
  flash_attn_kernel<<<BATCH * N_HEADS * (SEQ / 128), 256, 0, stream>>>(Qh, Kh, Vh, AO);

  // 5) Output projection -> fp32 d_out [B,S,H]
  gemm_wmma_kernel<<<dim3(HID / 128, MROWS / 128), 256, 0, stream>>>(
      AO, WoT, MROWS, HID, HID, 0, (float*)d_out, nullptr, 0);
}
